// GriffinBlock_58351425683813
// MI455X (gfx1250) — matrix operand order repaired and checked by reading, not compiled, not run on hardware
//
#include <hip/hip_runtime.h>
#include <hip/hip_bf16.h>

#define DEVFN __device__ __forceinline__

typedef __attribute__((ext_vector_type(16))) __bf16 v16bf;
typedef __attribute__((ext_vector_type(8)))  float  v8f;
typedef __attribute__((ext_vector_type(4)))  unsigned int u32x4;
typedef __attribute__((ext_vector_type(4)))  int v4i;

constexpr int kB = 4, kT = 2048, kDIM = 1024, kHID = 1536, kGH = 2048;
constexpr int kK = 4, kHD = 128, kQH = 8, kW = 1024;
constexpr int kM = kB * kT;          // 8192 rows
constexpr int kTP = kT + 64;         // padded key length
constexpr int kLD = 40;              // padded LDS row pitch (halfwords) for 32-K tiles

union TileU { v16bf v; u32x4 q[2]; };

#define WMMA_BF16(a, b, c) \
  __builtin_amdgcn_wmma_f32_16x16x32_bf16(false, (a), false, (b), (short)0, (c), false, false)

// ---- async global->LDS staging (CDNA5 ASYNCcnt path), with safe fallback ----
#if __has_builtin(__builtin_amdgcn_global_load_async_to_lds_b128) && \
    __has_builtin(__builtin_amdgcn_s_wait_asynccnt)
#define GEMM_ASYNC_STAGE 1
#else
#define GEMM_ASYNC_STAGE 0
#endif

typedef __attribute__((address_space(1))) v4i* as1_v4i_ptr;  // global int4*
typedef __attribute__((address_space(3))) v4i* as3_v4i_ptr;  // LDS int4*

DEVFN void cp16_g2l(unsigned short* dst_lds, const unsigned short* src_glb) {
#if GEMM_ASYNC_STAGE
  __builtin_amdgcn_global_load_async_to_lds_b128(
      (as1_v4i_ptr)src_glb, (as3_v4i_ptr)dst_lds, 0, 0);
#else
  *(u32x4*)dst_lds = *(const u32x4*)src_glb;   // global_load_b128 + ds_store_b128
#endif
}
#if GEMM_ASYNC_STAGE
#define STAGE_WAIT(n) __builtin_amdgcn_s_wait_asynccnt(n)
#else
#define STAGE_WAIT(n)
#endif

DEVFN unsigned short f2bf(float f) {
  unsigned int u = __float_as_uint(f);
  u += 0x7FFFu + ((u >> 16) & 1u);      // round-to-nearest-even
  return (unsigned short)(u >> 16);
}
DEVFN float geluf(float x) { return 0.5f * x * (1.0f + erff(x * 0.70710678118f)); }
DEVFN float sigmf(float x) { return 1.0f / (1.0f + __expf(-x)); }

// A-operand tile: 16(M) x 32(K), source row-major MxK (ld in elements).
// lanes 0-15: M=lane, K pairs {kh..kh+7, 16+kh..16+kh+7}, kh = 8*(lane>=16)
DEVFN v16bf load_a(const unsigned short* base, int ld) {
  int lane = threadIdx.x & 31;
  const unsigned short* p = base + (size_t)(lane & 15) * ld + ((lane >> 4) << 3);
  TileU t;
  t.q[0] = *(const u32x4*)p;
  t.q[1] = *(const u32x4*)(p + 16);
  return t.v;
}
// B-operand tile: 32(K) x 16(N), source row-major NxK (i.e. B^T, ld in elements).
// lane: N=lane%16, K pairs {kh..kh+7, 16+kh..16+kh+7}, kh = 8*(lane>=16)
DEVFN v16bf load_b(const unsigned short* base, int ld) {
  int lane = threadIdx.x & 31;
  const unsigned short* p = base + (size_t)(lane & 15) * ld + ((lane >> 4) << 3);
  TileU t;
  t.q[0] = *(const u32x4*)p;
  t.q[1] = *(const u32x4*)(p + 16);
  return t.v;
}

// ---------------------------------------------------------------- GEMM
// C[M x N] = A[M x K](bf16) * W[N x K]^T(bf16) + bias
// Block = 8 waves -> 128x64 macro tile; A/B staged in LDS, double buffered,
// async DMA of step k+32 overlapped with WMMAs of step k.
__global__ void __launch_bounds__(256) k_gemm(const unsigned short* __restrict__ A,
                                              const unsigned short* __restrict__ Wm,
                                              const float* __restrict__ bias,
                                              float* __restrict__ C, int N, int K) {
  __shared__ alignas(16) unsigned short sA[2][128 * kLD];
  __shared__ alignas(16) unsigned short sB[2][64 * kLD];
  const int tid  = threadIdx.x;
  const int wave = tid >> 5;
  const int ntile = blockIdx.x;
  const int m0 = blockIdx.y * 128;

  const int rB  = tid >> 2;            // 0..63
  const int kq  = (tid & 3) * 8;       // 0,8,16,24
  const int rA1 = rB + 64;             // 64..127

  const unsigned short* Wbase = Wm + (size_t)(ntile * 64 + rB) * K + kq;
  const unsigned short* Abase0 = A + (size_t)(m0 + rB)  * K + kq;
  const unsigned short* Abase1 = A + (size_t)(m0 + rA1) * K + kq;
  unsigned short* dB  = &sB[0][rB  * kLD + kq];
  unsigned short* dA0 = &sA[0][rB  * kLD + kq];
  unsigned short* dA1 = &sA[0][rA1 * kLD + kq];
  const size_t bufStrideB = 64 * kLD, bufStrideA = 128 * kLD;

  auto stage = [&](int buf, int k) {
    cp16_g2l(dB  + buf * bufStrideB, Wbase  + k);
    cp16_g2l(dA0 + buf * bufStrideA, Abase0 + k);
    cp16_g2l(dA1 + buf * bufStrideA, Abase1 + k);
  };

  v8f acc[4] = {};
  stage(0, 0);
  int buf = 0;
  for (int k = 0; k < K; k += 32) {
    if (k + 32 < K) {
      stage(buf ^ 1, k + 32);
      STAGE_WAIT(3);                 // drain current buffer, leave next in flight
    } else {
      STAGE_WAIT(0);
    }
    __syncthreads();
    v16bf a = load_a(&sA[buf][(wave * 16) * kLD], kLD);
#pragma unroll
    for (int j = 0; j < 4; ++j) {
      v16bf b = load_b(&sB[buf][(j * 16) * kLD], kLD);
      acc[j] = WMMA_BF16(a, b, acc[j]);
    }
    __syncthreads();                 // readers done before buf is re-staged
    buf ^= 1;
  }

  int lane = threadIdx.x & 31, n = lane & 15, mb = (lane >> 4) << 3;
  float* Cb = C + (size_t)(m0 + wave * 16) * N + ntile * 64;
#pragma unroll
  for (int j = 0; j < 4; ++j) {
    float bv = bias ? bias[ntile * 64 + j * 16 + n] : 0.0f;
#pragma unroll
    for (int v = 0; v < 8; ++v)
      Cb[(size_t)(mb + v) * N + j * 16 + n] = acc[j][v] + bv;
  }
}

// ---------------------------------------------------------------- elementwise
__global__ void k_cvt_bf16(const float* __restrict__ s, unsigned short* __restrict__ d, size_t n) {
  size_t i = (size_t)blockIdx.x * 256 + threadIdx.x;
  if (i < n) d[i] = f2bf(s[i]);
}
__global__ void k_zero_u16(unsigned short* __restrict__ d, size_t n) {
  size_t i = (size_t)blockIdx.x * 256 + threadIdx.x;
  if (i < n) d[i] = 0;
}
__global__ void k_add(float* __restrict__ X, const float* __restrict__ Y, size_t n) {
  size_t i = (size_t)blockIdx.x * 256 + threadIdx.x;
  if (i < n) X[i] += Y[i];
}
__global__ void k_add_out(const float* __restrict__ X, const float* __restrict__ Y,
                          float* __restrict__ O, size_t n) {
  size_t i = (size_t)blockIdx.x * 256 + threadIdx.x;
  if (i < n) O[i] = X[i] + Y[i];
}

// rmsnorm: one block (256 thr) per row of 1024
__global__ void __launch_bounds__(256) k_rmsnorm(const float* __restrict__ X,
                                                 const float* __restrict__ g,
                                                 unsigned short* __restrict__ XN) {
  int row = blockIdx.x;
  const float* xr = X + (size_t)row * kDIM;
  float ss = 0.f;
  for (int i = threadIdx.x; i < kDIM; i += 256) { float v = xr[i]; ss += v * v; }
  __shared__ float red[8];
  for (int o = 16; o > 0; o >>= 1) ss += __shfl_xor(ss, o, 32);
  if ((threadIdx.x & 31) == 0) red[threadIdx.x >> 5] = ss;
  __syncthreads();
  if (threadIdx.x < 8) {
    float v = red[threadIdx.x];
    for (int o = 4; o > 0; o >>= 1) v += __shfl_xor(v, o, 8);
    if (threadIdx.x == 0) red[0] = v;
  }
  __syncthreads();
  float inv = rsqrtf(red[0]) * 32.0f;   // sqrt(DIM)=32
  unsigned short* out = XN + (size_t)row * kDIM;
  for (int i = threadIdx.x; i < kDIM; i += 256) out[i] = f2bf(xr[i] * inv * g[i]);
}

// causal depthwise conv K=4 over time; input = h half of hawk grow output
__global__ void k_conv(const float* __restrict__ GHbuf, const float* __restrict__ cw,
                       const float* __restrict__ cb, float* __restrict__ HF,
                       unsigned short* __restrict__ HB) {
  size_t tid = (size_t)blockIdx.x * 256 + threadIdx.x;
  if (tid >= (size_t)kM * kHID) return;
  int c = (int)(tid % kHID);
  size_t row = tid / kHID;
  int t = (int)(row & (kT - 1));
  float acc = cb[c];
#pragma unroll
  for (int j = 0; j < kK; ++j) {
    int ts = t + j - (kK - 1);
    if (ts >= 0)
      acc += cw[c * kK + j] * GHbuf[(row + (size_t)(ts - t)) * (2 * kHID) + kHID + c];
  }
  HF[tid] = acc;
  HB[tid] = f2bf(acc);
}

__global__ void k_hawk_gates(float* __restrict__ FG, const float* __restrict__ HF,
                             const float* __restrict__ fb) {
  size_t tid = (size_t)blockIdx.x * 256 + threadIdx.x;
  if (tid >= (size_t)kM * kHID) return;
  int c = (int)(tid % kHID);
  size_t row = tid / kHID;
  float f  = FG[row * (2 * kHID) + c];
  float ip = FG[row * (2 * kHID) + kHID + c];
  float sp = log1pf(__expf(fb[c]));
  float alpha = __expf(-8.0f * sp * sigmf(f));
  float beta  = sqrtf(1.0f - alpha * alpha + 1e-6f);
  float xs = beta * sigmf(ip) * HF[tid];
  FG[row * (2 * kHID) + c] = alpha;
  FG[row * (2 * kHID) + kHID + c] = xs;
}

// sequential linear recurrence per (batch, channel)
__global__ void k_scan(float* __restrict__ FG) {
  int tid = blockIdx.x * 256 + threadIdx.x;
  if (tid >= kB * kHID) return;
  int c = tid % kHID, b = tid / kHID;
  float h = 0.f;
  float* base = FG + (size_t)b * kT * (2 * kHID);
  for (int t = 0; t < kT; ++t) {
    float* p = base + (size_t)t * (2 * kHID);
    h = p[c] * h + p[kHID + c];
    p[kHID + c] = h;
  }
}

__global__ void k_hawk_mix(const float* __restrict__ GHbuf, const float* __restrict__ FG,
                           unsigned short* __restrict__ HB) {
  size_t tid = (size_t)blockIdx.x * 256 + threadIdx.x;
  if (tid >= (size_t)kM * kHID) return;
  int c = (int)(tid % kHID);
  size_t row = tid / kHID;
  float g  = GHbuf[row * (2 * kHID) + c];
  float hs = FG[row * (2 * kHID) + kHID + c];
  HB[tid] = f2bf(geluf(g) * hs);
}

__global__ void k_mlp_act(const float* __restrict__ GHbuf, unsigned short* __restrict__ HB) {
  size_t tid = (size_t)blockIdx.x * 256 + threadIdx.x;
  if (tid >= (size_t)kM * kGH) return;
  int c = (int)(tid % kGH);
  size_t row = tid / kGH;
  float g = GHbuf[row * (2 * kGH) + c];
  float h = GHbuf[row * (2 * kGH) + kGH + c];
  HB[tid] = f2bf(geluf(g) * h);
}

// rotary + bf16 conversion: Q -> (b,t,h,d); K -> padded (b,t,d); V -> transposed (b,d,t)
__global__ void k_qkv_prep(const float* __restrict__ Qf, const float* __restrict__ KVf,
                           unsigned short* __restrict__ Qb, unsigned short* __restrict__ Kb,
                           unsigned short* __restrict__ Vt) {
  unsigned int tid = blockIdx.x * 256 + threadIdx.x;   // < 2^22
  int p = tid & 63;
  int h = (tid >> 6) & 7;
  int t = (tid >> 9) & (kT - 1);
  int b = tid >> 20;
  float inv = __expf(-(float)p * 0.14391565f);  // ln(10000)/64
  float ang = (float)t * inv;
  float cs = cosf(ang), sn = sinf(ang);
  size_t row = (size_t)b * kT + t;
  float q0 = Qf[row * (kQH * kHD) + h * kHD + 2 * p];
  float q1 = Qf[row * (kQH * kHD) + h * kHD + 2 * p + 1];
  size_t qo = row * (kQH * kHD) + h * kHD + 2 * p;
  Qb[qo]     = f2bf(q0 * cs - q1 * sn);
  Qb[qo + 1] = f2bf(q1 * cs + q0 * sn);
  if (h == 0) {
    float k0 = KVf[row * (2 * kHD) + 2 * p];
    float k1 = KVf[row * (2 * kHD) + 2 * p + 1];
    size_t ko = ((size_t)b * kTP + t) * kHD + 2 * p;
    Kb[ko]     = f2bf(k0 * cs - k1 * sn);
    Kb[ko + 1] = f2bf(k1 * cs + k0 * sn);
    float v0 = KVf[row * (2 * kHD) + kHD + 2 * p];
    float v1 = KVf[row * (2 * kHD) + kHD + 2 * p + 1];
    Vt[((size_t)b * kHD + 2 * p) * kTP + t]     = f2bf(v0);
    Vt[((size_t)b * kHD + 2 * p + 1) * kTP + t] = f2bf(v1);
  }
}

// flash sliding-window attention: 1 wave per (b, head, 16-query tile)
__global__ void __launch_bounds__(128) k_attn(const unsigned short* __restrict__ Q,
                                              const unsigned short* __restrict__ Kb,
                                              const unsigned short* __restrict__ Vt,
                                              unsigned short* __restrict__ O) {
  __shared__ alignas(16) unsigned short psh[4][16 * 32];
  int wid = threadIdx.x >> 5, lane = threadIdx.x & 31;
  int tile = blockIdx.x * 4 + wid;       // 4096 tiles total
  int qt = tile & 127;
  int h  = (tile >> 7) & 7;
  int b  = tile >> 10;
  int q0 = qt * 16;
  const unsigned short* Qbase = Q + ((size_t)b * kT + q0) * (kQH * kHD) + h * kHD;
  v16bf qa0 = load_a(Qbase,      kQH * kHD);
  v16bf qa1 = load_a(Qbase + 32, kQH * kHD);
  v16bf qa2 = load_a(Qbase + 64, kQH * kHD);
  v16bf qa3 = load_a(Qbase + 96, kQH * kHD);
  v8f o[8] = {};
  float rm[8], rl[8];
#pragma unroll
  for (int v = 0; v < 8; ++v) { rm[v] = -1e30f; rl[v] = 0.f; }
  const unsigned short* Kbb = Kb + (size_t)b * kTP * kHD;
  const unsigned short* Vbb = Vt + (size_t)b * kHD * kTP;
  int n = lane & 15, mb = (lane >> 4) << 3;
  int ks0 = q0 - kW; if (ks0 < 0) ks0 = 0;
  const float scale = 0.08838834764f;    // 1/sqrt(128)
  unsigned short* pw = &psh[wid][0];
  for (int kc = ks0; kc < q0 + 16; kc += 32) {
    v8f s0 = {}, s1 = {};
    const unsigned short* K0 = Kbb + (size_t)kc * kHD;
    const unsigned short* K1 = Kbb + (size_t)(kc + 16) * kHD;
    s0 = WMMA_BF16(qa0, load_b(K0,      kHD), s0);
    s1 = WMMA_BF16(qa0, load_b(K1,      kHD), s1);
    s0 = WMMA_BF16(qa1, load_b(K0 + 32, kHD), s0);
    s1 = WMMA_BF16(qa1, load_b(K1 + 32, kHD), s1);
    s0 = WMMA_BF16(qa2, load_b(K0 + 64, kHD), s0);
    s1 = WMMA_BF16(qa2, load_b(K1 + 64, kHD), s1);
    s0 = WMMA_BF16(qa3, load_b(K0 + 96, kHD), s0);
    s1 = WMMA_BF16(qa3, load_b(K1 + 96, kHD), s1);
#pragma unroll
    for (int v = 0; v < 8; ++v) {
      int qp = q0 + mb + v;
      int kp0 = kc + n, kp1 = kc + 16 + n;
      float x0 = (kp0 <= qp && kp0 >= qp - kW) ? s0[v] * scale : -1e30f;
      float x1 = (kp1 <= qp && kp1 >= qp - kW) ? s1[v] * scale : -1e30f;
      float mx = fmaxf(x0, x1);
      for (int ofs = 8; ofs > 0; ofs >>= 1) mx = fmaxf(mx, __shfl_xor(mx, ofs, 16));
      float mnew = fmaxf(rm[v], mx);
      float corr = __expf(rm[v] - mnew);
      float p0 = __expf(x0 - mnew), p1 = __expf(x1 - mnew);
      float rs = p0 + p1;
      for (int ofs = 8; ofs > 0; ofs >>= 1) rs += __shfl_xor(rs, ofs, 16);
      rm[v] = mnew;
      rl[v] = rl[v] * corr + rs;
#pragma unroll
      for (int j = 0; j < 8; ++j) o[j][v] = o[j][v] * corr;
      pw[(mb + v) * 32 + n]      = f2bf(p0);
      pw[(mb + v) * 32 + 16 + n] = f2bf(p1);
    }
    v16bf pa = load_a(pw, 32);
#pragma unroll
    for (int j = 0; j < 8; ++j) {
      v16bf vb = load_b(Vbb + (size_t)(j * 16) * kTP + kc, kTP);
      o[j] = WMMA_BF16(pa, vb, o[j]);
    }
  }
  unsigned short* Ob = O + ((size_t)b * kT + q0) * (kQH * kHD) + h * kHD;
#pragma unroll
  for (int j = 0; j < 8; ++j)
#pragma unroll
    for (int v = 0; v < 8; ++v)
      Ob[(size_t)(mb + v) * (kQH * kHD) + j * 16 + n] = f2bf(o[j][v] / rl[v]);
}

// ---------------------------------------------------------------- host
static inline unsigned int gblk(size_t n) { return (unsigned int)((n + 255) / 256); }

extern "C" void kernel_launch(void* const* d_in, const int* in_sizes, int n_in,
                              void* d_out, int out_size, void* d_ws, size_t ws_size,
                              hipStream_t stream) {
  const float* x        = (const float*)d_in[0];
  const float* gn_hawk  = (const float*)d_in[1];
  const float* gn_hgmlp = (const float*)d_in[2];
  const float* gn_smqa  = (const float*)d_in[3];
  const float* gn_sgmlp = (const float*)d_in[4];
  const float* hawk_Win = (const float*)d_in[5];
  const float* hawk_cw  = (const float*)d_in[6];
  const float* hawk_cb  = (const float*)d_in[7];
  const float* hawk_Wg  = (const float*)d_in[8];
  const float* hawk_bg  = (const float*)d_in[9];
  const float* hawk_fb  = (const float*)d_in[10];
  const float* hawk_Wout= (const float*)d_in[11];
  const float* g1_Wgrow = (const float*)d_in[12];
  const float* g1_Wshr  = (const float*)d_in[13];
  const float* g2_Wgrow = (const float*)d_in[14];
  const float* g2_Wshr  = (const float*)d_in[15];
  const float* smqa_Wq  = (const float*)d_in[16];
  const float* smqa_Wkv = (const float*)d_in[17];
  const float* smqa_Wo  = (const float*)d_in[18];
  float* out = (float*)d_out;

  char* ws = (char*)d_ws;
  size_t off = 0;
  auto alloc = [&](size_t bytes) -> char* {
    char* p = ws + off;
    off += (bytes + 255) & ~(size_t)255;
    return p;
  };
  auto allocU = [&](size_t elems) { return (unsigned short*)alloc(elems * 2); };
  auto allocF = [&](size_t elems) { return (float*)alloc(elems * 4); };

  // bf16 weights
  unsigned short* wWin  = allocU((size_t)2 * kHID * kDIM);
  unsigned short* wWg   = allocU((size_t)2 * kHID * kHID);
  unsigned short* wWout = allocU((size_t)kDIM * kHID);
  unsigned short* wG1g  = allocU((size_t)2 * kGH * kDIM);
  unsigned short* wG1s  = allocU((size_t)kDIM * kGH);
  unsigned short* wG2g  = allocU((size_t)2 * kGH * kDIM);
  unsigned short* wG2s  = allocU((size_t)kDIM * kGH);
  unsigned short* wWq   = allocU((size_t)kQH * kHD * kDIM);
  unsigned short* wWkv  = allocU((size_t)2 * kHD * kDIM);
  unsigned short* wWo   = allocU((size_t)kDIM * kDIM);
  // activations
  float*          X    = allocF((size_t)kM * kDIM);
  unsigned short* XN   = allocU((size_t)kM * kDIM);
  float*          Y    = allocF((size_t)kM * kDIM);
  float*          BUF1 = allocF((size_t)kM * 2 * kGH);   // largest grow output
  float*          BUF2 = allocF((size_t)kM * 2 * kHID);  // hawk forget/input
  float*          HF   = allocF((size_t)kM * kHID);      // conv output fp32
  unsigned short* HB   = allocU((size_t)kM * kGH);       // bf16 hidden (M x 2048 max)
  float*          KVF  = allocF((size_t)kM * 2 * kHD);
  unsigned short* Qb   = allocU((size_t)kM * kQH * kHD);
  unsigned short* Kb   = allocU((size_t)kB * kTP * kHD);
  unsigned short* Vt   = allocU((size_t)kB * kHD * kTP);

  auto cvt = [&](const float* s, unsigned short* d, size_t n) {
    k_cvt_bf16<<<gblk(n), 256, 0, stream>>>(s, d, n);
  };
  auto gemm = [&](const unsigned short* A, const unsigned short* Wm, const float* bias,
                  float* C, int N, int K) {
    dim3 g(N / 64, kM / 128);
    k_gemm<<<g, 256, 0, stream>>>(A, Wm, bias, C, N, K);
  };

  cvt(hawk_Win,  wWin,  (size_t)2 * kHID * kDIM);
  cvt(hawk_Wg,   wWg,   (size_t)2 * kHID * kHID);
  cvt(hawk_Wout, wWout, (size_t)kDIM * kHID);
  cvt(g1_Wgrow,  wG1g,  (size_t)2 * kGH * kDIM);
  cvt(g1_Wshr,   wG1s,  (size_t)kDIM * kGH);
  cvt(g2_Wgrow,  wG2g,  (size_t)2 * kGH * kDIM);
  cvt(g2_Wshr,   wG2s,  (size_t)kDIM * kGH);
  cvt(smqa_Wq,   wWq,   (size_t)kQH * kHD * kDIM);
  cvt(smqa_Wkv,  wWkv,  (size_t)2 * kHD * kDIM);
  cvt(smqa_Wo,   wWo,   (size_t)kDIM * kDIM);

  const size_t nXD = (size_t)kM * kDIM;
  (void)hipMemcpyAsync(X, x, nXD * sizeof(float), hipMemcpyDeviceToDevice, stream);

  // ---- Hawk block ----
  k_rmsnorm<<<kM, 256, 0, stream>>>(X, gn_hawk, XN);
  gemm(XN, wWin, nullptr, BUF1, 2 * kHID, kDIM);
  k_conv<<<gblk((size_t)kM * kHID), 256, 0, stream>>>(BUF1, hawk_cw, hawk_cb, HF, HB);
  gemm(HB, wWg, hawk_bg, BUF2, 2 * kHID, kHID);
  k_hawk_gates<<<gblk((size_t)kM * kHID), 256, 0, stream>>>(BUF2, HF, hawk_fb);
  k_scan<<<gblk((size_t)kB * kHID), 256, 0, stream>>>(BUF2);
  k_hawk_mix<<<gblk((size_t)kM * kHID), 256, 0, stream>>>(BUF1, BUF2, HB);
  gemm(HB, wWout, nullptr, Y, kDIM, kHID);
  k_add<<<gblk(nXD), 256, 0, stream>>>(X, Y, nXD);

  // ---- Gated MLP 1 ----
  k_rmsnorm<<<kM, 256, 0, stream>>>(X, gn_hgmlp, XN);
  gemm(XN, wG1g, nullptr, BUF1, 2 * kGH, kDIM);
  k_mlp_act<<<gblk((size_t)kM * kGH), 256, 0, stream>>>(BUF1, HB);
  gemm(HB, wG1s, nullptr, Y, kDIM, kGH);
  k_add<<<gblk(nXD), 256, 0, stream>>>(X, Y, nXD);

  // ---- Sliding-window MQA ----
  k_rmsnorm<<<kM, 256, 0, stream>>>(X, gn_smqa, XN);
  gemm(XN, wWq, nullptr, BUF1, kQH * kHD, kDIM);
  gemm(XN, wWkv, nullptr, KVF, 2 * kHD, kDIM);
  {
    size_t npad = (size_t)kB * kTP * kHD;
    k_zero_u16<<<gblk(npad), 256, 0, stream>>>(Kb, npad);
    k_zero_u16<<<gblk(npad), 256, 0, stream>>>(Vt, npad);
  }
  k_qkv_prep<<<gblk((size_t)kB * kT * kQH * 64), 256, 0, stream>>>(BUF1, KVF, Qb, Kb, Vt);
  k_attn<<<(kB * kQH * (kT / 16)) / 4, 128, 0, stream>>>(Qb, Kb, Vt, HB);
  gemm(HB, wWo, nullptr, Y, kDIM, kDIM);
  k_add<<<gblk(nXD), 256, 0, stream>>>(X, Y, nXD);

  // ---- Gated MLP 2 (writes d_out) ----
  k_rmsnorm<<<kM, 256, 0, stream>>>(X, gn_sgmlp, XN);
  gemm(XN, wG2g, nullptr, BUF1, 2 * kGH, kDIM);
  k_mlp_act<<<gblk((size_t)kM * kGH), 256, 0, stream>>>(BUF1, HB);
  gemm(HB, wG2s, nullptr, Y, kDIM, kGH);
  k_add_out<<<gblk(nXD), 256, 0, stream>>>(X, Y, out, nXD);

  (void)in_sizes; (void)n_in; (void)out_size; (void)ws_size;
}
